// GRUModel_18657337934217
// MI455X (gfx1250) — matrix operand order repaired and checked by reading, not compiled, not run on hardware
//
#include <hip/hip_runtime.h>
#include <hip/hip_bf16.h>
#include <stdint.h>

// ---------------- problem constants ----------------
#define B_   32768
#define T_   62
#define I_   5
#define H_   16
#define K_   992          // T*H
#define F_   512
#define C_   3
#define KT_  31           // K_/32

// ---------------- vector types ----------------
typedef __attribute__((ext_vector_type(8)))  __bf16 v8bf;
typedef __attribute__((ext_vector_type(16))) __bf16 v16bf;
typedef __attribute__((ext_vector_type(8)))  float  v8f;
typedef __attribute__((ext_vector_type(4)))  unsigned int u32x4;
typedef __attribute__((ext_vector_type(8)))  int    i32x8;
typedef __attribute__((ext_vector_type(4)))  int    i32x4;

// ---------------- helpers ----------------
__device__ __forceinline__ unsigned short f2bf_rne(float f) {
    unsigned u = __float_as_uint(f);
    unsigned lsb = (u >> 16) & 1u;
    u += 0x7fffu + lsb;
    return (unsigned short)(u >> 16);
}
__device__ __forceinline__ unsigned pack_bf16x2(float lo, float hi) {
    return (unsigned)f2bf_rne(lo) | ((unsigned)f2bf_rne(hi) << 16);
}
__device__ __forceinline__ float sigm_f(float x) {
    x = fminf(fmaxf(x, -30.f), 30.f);
    float e = __builtin_amdgcn_exp2f(-1.44269504089f * x);
    return __builtin_amdgcn_rcpf(1.0f + e);
}
__device__ __forceinline__ float tanh_f(float x) {
    x = fminf(fmaxf(x, -15.f), 15.f);
    float e = __builtin_amdgcn_exp2f(2.88539008178f * x);   // e^(2x)
    return (e - 1.0f) * __builtin_amdgcn_rcpf(e + 1.0f);
}

// ---------------- kernel 1: W1 f32 -> bf16 ----------------
__global__ __launch_bounds__(256) void prep_w1_kernel(const float* __restrict__ W1,
                                                      uint16_t* __restrict__ w1bf) {
    int i = blockIdx.x * 256 + threadIdx.x;
    if (i < F_ * K_) w1bf[i] = f2bf_rne(W1[i]);
}

// ---------------- kernel 2: zero BN accumulators ----------------
__global__ __launch_bounds__(256) void zero_stats_kernel(float* __restrict__ s) {
    int i = blockIdx.x * 256 + threadIdx.x;
    if (i < 2 * F_) s[i] = 0.0f;
}

// ---------------- kernel 3: GRU scan ----------------
// one thread per batch row; h[16] lives in VGPRs; weights broadcast from LDS.
// Emits h as bf16 row-major (B, 992) — exactly the A-operand source for FC1.
__global__ __launch_bounds__(256) void gru_kernel(const float* __restrict__ x,
                                                  const float* __restrict__ w_ih,
                                                  const float* __restrict__ w_hh,
                                                  const float* __restrict__ b_ih,
                                                  const float* __restrict__ b_hh,
                                                  uint16_t* __restrict__ hbuf) {
    __shared__ float s_wih[3 * H_ * I_];   // 240
    __shared__ float s_whh[3 * H_ * H_];   // 768
    __shared__ float s_bih[3 * H_];        // 48
    __shared__ float s_bhh[3 * H_];        // 48
    const int tid = threadIdx.x;
    for (int i = tid; i < 240; i += 256) s_wih[i] = w_ih[i];
    for (int i = tid; i < 768; i += 256) s_whh[i] = w_hh[i];
    if (tid < 48) { s_bih[tid] = b_ih[tid]; s_bhh[tid] = b_hh[tid]; }
    __syncthreads();

    const int b = blockIdx.x * 256 + tid;
    const float* xr = x + (size_t)b * (T_ * I_);
    uint16_t* hb = hbuf + (size_t)b * K_;

    float h[H_];
#pragma unroll
    for (int j = 0; j < H_; ++j) h[j] = 0.0f;

    for (int t = 0; t < T_; ++t) {
        float xv[I_];
#pragma unroll
        for (int i = 0; i < I_; ++i) xv[i] = xr[t * I_ + i];

        float hn[H_];
#pragma unroll
        for (int j = 0; j < H_; ++j) {
            float gr = s_bih[j], gz = s_bih[H_ + j], gn = s_bih[2 * H_ + j];
#pragma unroll
            for (int i = 0; i < I_; ++i) {
                gr = fmaf(s_wih[j * I_ + i],            xv[i], gr);
                gz = fmaf(s_wih[(H_ + j) * I_ + i],     xv[i], gz);
                gn = fmaf(s_wih[(2 * H_ + j) * I_ + i], xv[i], gn);
            }
            float hr = s_bhh[j], hz = s_bhh[H_ + j], hh = s_bhh[2 * H_ + j];
#pragma unroll
            for (int k = 0; k < H_; ++k) {
                hr = fmaf(s_whh[j * H_ + k],            h[k], hr);
                hz = fmaf(s_whh[(H_ + j) * H_ + k],     h[k], hz);
                hh = fmaf(s_whh[(2 * H_ + j) * H_ + k], h[k], hh);
            }
            float r = sigm_f(gr + hr);
            float z = sigm_f(gz + hz);
            float n = tanh_f(gn + r * hh);
            hn[j] = (1.0f - z) * n + z * h[j];
        }
        unsigned* dst = (unsigned*)(hb + t * H_);
#pragma unroll
        for (int j = 0; j < 8; ++j) {
            h[2 * j]     = hn[2 * j];
            h[2 * j + 1] = hn[2 * j + 1];
            dst[j] = pack_bf16x2(hn[2 * j], hn[2 * j + 1]);
        }
    }
}

// ---------------- kernel 4: FC1 via WMMA bf16 + fused BN stats ----------------
// block = 128 threads (4 waves). Block tile: M=16 batch rows x N=512 features.
// A strip (16 x 992 bf16 = 31744B, contiguous in hbuf) staged into LDS with one
// TDM tensor_load_to_lds; each wave covers 128 features (8 n-tiles), K loop of 31.
__global__ __launch_bounds__(128) void fc1_wmma_kernel(const uint16_t* __restrict__ hbuf,
                                                       const uint16_t* __restrict__ w1bf,
                                                       const float* __restrict__ b1,
                                                       float* __restrict__ pre,
                                                       float* __restrict__ colsum,
                                                       float* __restrict__ colsq) {
    __shared__ __align__(128) uint16_t sA[16 * K_];   // 31744 B
    const int tid  = threadIdx.x;
    const int lane = tid & 31;
    const int wave = tid >> 5;
    const int row0 = blockIdx.x * 16;

#if defined(__has_builtin) && __has_builtin(__builtin_amdgcn_tensor_load_to_lds)
    if (wave == 0) {
        // D# group0: count=1 | lds_addr | global_addr(57b) | type=2
        unsigned lds_base = (unsigned)(size_t)(&sA[0]);
        unsigned long long ga =
            (unsigned long long)(size_t)hbuf + (unsigned long long)row0 * (K_ * 2ull);
        u32x4 g0;
        g0[0] = 1u;
        g0[1] = lds_base;
        g0[2] = (unsigned)(ga & 0xffffffffull);
        g0[3] = (unsigned)((ga >> 32) & 0x1ffffffull) | (2u << 30);
        // D# group1: data_size=4B; tensor 496 dwords x 32768 rows; tile 496 x 16
        const unsigned td0 = K_ / 2, td1 = B_, tl0 = K_ / 2, tl1 = 16, st0 = K_ / 2;
        i32x8 g1;
        g1[0] = (int)(2u << 16);                                   // data_size = 4B
        g1[1] = (int)((td0 & 0xffffu) << 16);                      // tensor_dim0 lo
        g1[2] = (int)((td0 >> 16) | ((td1 & 0xffffu) << 16));      // dim0 hi | dim1 lo
        g1[3] = (int)((td1 >> 16) | (tl0 << 16));                  // dim1 hi | tile_dim0
        g1[4] = (int)tl1;                                          // tile_dim1, tile_dim2=0
        g1[5] = (int)st0;                                          // tensor_dim0_stride lo
        g1[6] = 0;
        g1[7] = 0;
        i32x4 g2 = {0, 0, 0, 0}, g3 = {0, 0, 0, 0};
        i32x8 g4 = {0, 0, 0, 0, 0, 0, 0, 0};   // extra group on clang-23 toolchain
        __builtin_amdgcn_tensor_load_to_lds(g0, g1, g2, g3, g4, 0);
        __builtin_amdgcn_s_wait_tensorcnt(0);
    }
#else
    {   // fallback: the 16-row strip is contiguous in hbuf
        const v8bf* src = (const v8bf*)(hbuf + (size_t)row0 * K_);
        v8bf* dst = (v8bf*)sA;
        for (int i = tid; i < (16 * K_) / 8; i += 128) dst[i] = src[i];
    }
#endif
    __syncthreads();

    const int m  = lane & 15;
    const int hi = lane >> 4;            // 0: K 0..7/16..23, 1: K 8..15/24..31
    const int nbase = wave * 128;

    v8f c[8];
#pragma unroll
    for (int nt = 0; nt < 8; ++nt) {
        float bias = b1[nbase + nt * 16 + m];
#pragma unroll
        for (int r = 0; r < 8; ++r) c[nt][r] = bias;
    }

    const int aoff = m * K_ + hi * 8;
#pragma unroll 1
    for (int k0 = 0; k0 < KT_; ++k0) {
        const int k = k0 * 32;
        const uint16_t* ap = &sA[aoff + k];
        v8bf a0 = *(const v8bf*)(ap);        // K = hi*8 + 0..7
        v8bf a1 = *(const v8bf*)(ap + 16);   // K = hi*8 + 16..23
        v16bf a = __builtin_shufflevector(a0, a1, 0, 1, 2, 3, 4, 5, 6, 7,
                                                  8, 9, 10, 11, 12, 13, 14, 15);
#pragma unroll
        for (int nt = 0; nt < 8; ++nt) {
            const int n = nbase + nt * 16 + m;         // N = lane&15
            const uint16_t* bq = w1bf + (size_t)n * K_ + k + hi * 8;  // K = hi*8 + 0..7, 16 + hi*8 + 0..7
            v8bf b0 = *(const v8bf*)(bq);
            v8bf b1v = *(const v8bf*)(bq + 16);
            v16bf bmat = __builtin_shufflevector(b0, b1v, 0, 1, 2, 3, 4, 5, 6, 7,
                                                 8, 9, 10, 11, 12, 13, 14, 15);
            c[nt] = __builtin_amdgcn_wmma_f32_16x16x32_bf16(
                false, a, false, bmat, (short)0, c[nt], false, false);
        }
    }

    // epilogue: store pre tile, fuse per-column sum / sum-of-squares
#pragma unroll
    for (int nt = 0; nt < 8; ++nt) {
        const int col = nbase + nt * 16 + m;
        float s1 = 0.0f, s2 = 0.0f;
#pragma unroll
        for (int r = 0; r < 8; ++r) {
            float v = c[nt][r];
            const int row = row0 + r + hi * 8;          // C/D layout
            pre[(size_t)row * F_ + col] = v;
            s1 += v;
            s2 += v * v;
        }
        s1 += __shfl_xor(s1, 16, 32);                   // lane L and L^16 share col
        s2 += __shfl_xor(s2, 16, 32);
        if (hi == 0) {
            atomicAdd(&colsum[col], s1);
            atomicAdd(&colsq[col],  s2);
        }
    }
}

// ---------------- kernel 5: BN(train) + ReLU + classifier ----------------
// one wave per batch row; lane owns 16 features. scale/shift + Wc cached in LDS.
__global__ __launch_bounds__(256) void head_kernel(const float* __restrict__ pre,
                                                   const float* __restrict__ colsum,
                                                   const float* __restrict__ colsq,
                                                   const float* __restrict__ gamma,
                                                   const float* __restrict__ beta,
                                                   const float* __restrict__ Wc,
                                                   const float* __restrict__ bc,
                                                   float* __restrict__ out_logits,
                                                   float* __restrict__ out_feat) {
    __shared__ float sScale[F_], sShift[F_], sWc[C_ * F_];
    const int tid = threadIdx.x;
    const float invB = 1.0f / (float)B_;
    for (int f = tid; f < F_; f += 256) {
        float mu  = colsum[f] * invB;
        float var = fmaf(-mu, mu, colsq[f] * invB);
        float rs  = rsqrtf(var + 1e-5f);
        float sc  = gamma[f] * rs;
        sScale[f] = sc;
        sShift[f] = fmaf(-mu, sc, beta[f]);
    }
    for (int i = tid; i < C_ * F_; i += 256) sWc[i] = Wc[i];
    __syncthreads();

    const int lane = tid & 31;
    const int wave = tid >> 5;
    const int row  = blockIdx.x * 8 + wave;

    const float4* prow = (const float4*)(pre + (size_t)row * F_ + lane * 16);
    float4* frow = (float4*)(out_feat + (size_t)row * F_ + lane * 16);

    float a0 = 0.f, a1 = 0.f, a2 = 0.f;
#pragma unroll
    for (int q = 0; q < 4; ++q) {
        float4 p = prow[q];
        const int f = lane * 16 + q * 4;
        float4 v;
        v.x = fmaxf(0.f, fmaf(p.x, sScale[f + 0], sShift[f + 0]));
        v.y = fmaxf(0.f, fmaf(p.y, sScale[f + 1], sShift[f + 1]));
        v.z = fmaxf(0.f, fmaf(p.z, sScale[f + 2], sShift[f + 2]));
        v.w = fmaxf(0.f, fmaf(p.w, sScale[f + 3], sShift[f + 3]));
        frow[q] = v;
        a0 += v.x * sWc[f] + v.y * sWc[f + 1] + v.z * sWc[f + 2] + v.w * sWc[f + 3];
        a1 += v.x * sWc[F_ + f] + v.y * sWc[F_ + f + 1] + v.z * sWc[F_ + f + 2] + v.w * sWc[F_ + f + 3];
        a2 += v.x * sWc[2 * F_ + f] + v.y * sWc[2 * F_ + f + 1] + v.z * sWc[2 * F_ + f + 2] + v.w * sWc[2 * F_ + f + 3];
    }
#pragma unroll
    for (int off = 16; off > 0; off >>= 1) {
        a0 += __shfl_xor(a0, off, 32);
        a1 += __shfl_xor(a1, off, 32);
        a2 += __shfl_xor(a2, off, 32);
    }
    if (lane == 0) {
        out_logits[(size_t)row * C_ + 0] = a0 + bc[0];
        out_logits[(size_t)row * C_ + 1] = a1 + bc[1];
        out_logits[(size_t)row * C_ + 2] = a2 + bc[2];
    }
}

// ---------------- workspace layout (bytes) ----------------
// hbuf  : bf16  B*992            = 65,011,712
// pre   : f32   B*512            = 67,108,864
// w1bf  : bf16  512*992          =  1,015,808
// stats : f32   2*512            =      4,096
static const size_t OFF_HBUF = 0;
static const size_t OFF_PRE  = OFF_HBUF + (size_t)B_ * K_ * 2;
static const size_t OFF_W1BF = OFF_PRE  + (size_t)B_ * F_ * 4;
static const size_t OFF_SUM  = OFF_W1BF + (size_t)F_ * K_ * 2;
static const size_t OFF_SQ   = OFF_SUM  + (size_t)F_ * 4;

extern "C" void kernel_launch(void* const* d_in, const int* in_sizes, int n_in,
                              void* d_out, int out_size, void* d_ws, size_t ws_size,
                              hipStream_t stream) {
    const float* x     = (const float*)d_in[0];
    const float* w_ih  = (const float*)d_in[1];
    const float* w_hh  = (const float*)d_in[2];
    const float* b_ih  = (const float*)d_in[3];
    const float* b_hh  = (const float*)d_in[4];
    const float* W1    = (const float*)d_in[5];
    const float* b1    = (const float*)d_in[6];
    const float* gamma = (const float*)d_in[7];
    const float* beta  = (const float*)d_in[8];
    const float* Wc    = (const float*)d_in[9];
    const float* bc    = (const float*)d_in[10];

    char* ws = (char*)d_ws;
    uint16_t* hbuf   = (uint16_t*)(ws + OFF_HBUF);
    float*    pre    = (float*)   (ws + OFF_PRE);
    uint16_t* w1bf   = (uint16_t*)(ws + OFF_W1BF);
    float*    colsum = (float*)   (ws + OFF_SUM);
    float*    colsq  = (float*)   (ws + OFF_SQ);

    float* out_logits = (float*)d_out;                        // (B, 3)
    float* out_feat   = (float*)d_out + (size_t)B_ * C_;      // (B, 512)

    prep_w1_kernel<<<(F_ * K_ + 255) / 256, 256, 0, stream>>>(W1, w1bf);
    zero_stats_kernel<<<(2 * F_ + 255) / 256, 256, 0, stream>>>(colsum);
    gru_kernel<<<B_ / 256, 256, 0, stream>>>(x, w_ih, w_hh, b_ih, b_hh, hbuf);
    fc1_wmma_kernel<<<B_ / 16, 128, 0, stream>>>(hbuf, w1bf, b1, pre, colsum, colsq);
    head_kernel<<<B_ / 8, 256, 0, stream>>>(pre, colsum, colsq, gamma, beta, Wc, bc,
                                            out_logits, out_feat);
}
